// Attention_9096740733536
// MI455X (gfx1250) — matrix operand order repaired and checked by reading, not compiled, not run on hardware
//
#include <hip/hip_runtime.h>
#include <hip/hip_bf16.h>
#include <math.h>
#include <stdint.h>

#define H_    8
#define D_    128
#define DIM_  1024
#define QB_   16
#define N_    512
#define ROWS_ 8192   // QB_*N_
#define HQ_   128    // H_*QB_

typedef __attribute__((ext_vector_type(16))) _Float16 v16h;
typedef __attribute__((ext_vector_type(8)))  _Float16 v8h;
typedef __attribute__((ext_vector_type(8)))  float    v8f;

__device__ __forceinline__ v8f wmma_f16(v16h a, v16h b, v8f c) {
    return __builtin_amdgcn_wmma_f32_16x16x32_f16(false, a, false, b, (short)0, c,
                                                  false, false);
}

// A fragment: 16x32 f16 tile, row-major, lda in halves.
// lanes 0-15: M=lane, K={0..7,16..23}; lanes 16-31: M=lane-16, K={8..15,24..31}
__device__ __forceinline__ v16h load_fragA(const _Float16* __restrict__ A, int lda, int lane) {
    int m  = lane & 15;
    int kb = (lane >> 4) * 8;
    const _Float16* p = A + (size_t)m * lda + kb;
    v8h lo = *(const v8h*)(p);
    v8h hi = *(const v8h*)(p + 16);
    return __builtin_shufflevector(lo, hi, 0,1,2,3,4,5,6,7,8,9,10,11,12,13,14,15);
}

// B fragment: 32x16 f16 tile given as Bt (row n = output column, K contiguous), ldb in halves.
// lanes 0-15: N=lane, K={0..7,16..23}; lanes 16-31: N=lane-16, K={8..15,24..31}
__device__ __forceinline__ v16h load_fragB(const _Float16* __restrict__ Bt, int ldb, int lane) {
    int nn = lane & 15;
    int ks = (lane >> 4) * 8;
    const _Float16* p = Bt + (size_t)nn * ldb + ks;
    v8h lo = *(const v8h*)(p);
    v8h hi = *(const v8h*)(p + 16);
    return __builtin_shufflevector(lo, hi, 0,1,2,3,4,5,6,7,8,9,10,11,12,13,14,15);
}

__device__ __forceinline__ v8f zero8() {
    v8f z = {0.f,0.f,0.f,0.f,0.f,0.f,0.f,0.f};
    return z;
}

// CDNA5 async global -> LDS copy (16B per lane), tracked by ASYNCcnt.
__device__ __forceinline__ void async_copy_b128(const void* gptr, void* lptr) {
    uint32_t lds_off = (uint32_t)(uintptr_t)lptr;   // low 32 bits of LDS aperture addr
    asm volatile("global_load_async_to_lds_b128 %0, %1, off"
                 :: "v"(lds_off), "v"((uintptr_t)gptr) : "memory");
}

// ---------------------------------------------------------------------------
// utility kernels
// ---------------------------------------------------------------------------
__global__ void zero_kernel(float* p, int n) {
    int i = blockIdx.x * blockDim.x + threadIdx.x;
    if (i < n) p[i] = 0.f;
}

__global__ void cvt_f32_f16_kernel(const float* __restrict__ src,
                                   _Float16* __restrict__ dst, int n) {
    int i = blockIdx.x * 256 + threadIdx.x;
    if (i < n) dst[i] = (_Float16)src[i];
}

// LayerNorm over rows of 1024, f32 in -> f16 out. One 256-thread block per row.
__global__ void ln_rows_kernel(const float* __restrict__ x,
                               const float* __restrict__ g,
                               const float* __restrict__ b,
                               _Float16* __restrict__ y) {
    __shared__ float red[2][8];
    int row = blockIdx.x;
    const float* xr = x + (size_t)row * DIM_;
    float vals[4];
    float s = 0.f, ss = 0.f;
#pragma unroll
    for (int i = 0; i < 4; i++) {
        float vv = xr[threadIdx.x + 256 * i];
        vals[i] = vv; s += vv; ss += vv * vv;
    }
#pragma unroll
    for (int off = 16; off > 0; off >>= 1) {
        s  += __shfl_xor(s,  off, 32);
        ss += __shfl_xor(ss, off, 32);
    }
    int wid = threadIdx.x >> 5;
    if ((threadIdx.x & 31) == 0) { red[0][wid] = s; red[1][wid] = ss; }
    __syncthreads();
    float ts = 0.f, tss = 0.f;
#pragma unroll
    for (int i = 0; i < 8; i++) { ts += red[0][i]; tss += red[1][i]; }
    float mu   = ts * (1.f / DIM_);
    float var  = tss * (1.f / DIM_) - mu * mu;
    float rstd = rsqrtf(var + 1e-5f);
    _Float16* yr = y + (size_t)row * DIM_;
#pragma unroll
    for (int i = 0; i < 4; i++) {
        int c = threadIdx.x + 256 * i;
        yr[c] = (_Float16)((vals[i] - mu) * rstd * g[c] + b[c]);
    }
}

// ---------------------------------------------------------------------------
// Block-tiled GEMM (M x 1024) @ (1024 x 1024)^T with async-LDS double buffering.
// Block = 256 thr (8 waves), block tile 128x128, K-panels of 32 staged in LDS.
// Wave grid 4x2: each wave does a 32x64 sub-tile = 2x4 WMMAs per K-step.
// EPI 0: f16 scatter dst[h][qb][n][d]    (q/k projection)
// EPI 1: f16 scatter dst[h][qb][d][n]    (v projection, transposed)
// EPI 2: f32 dst[r][c] = acc + bout[c]   (final projection)
// ---------------------------------------------------------------------------
#define LSTR 40   // 32 halves + 8 pad (80B row stride, 16B aligned)

template <int EPI>
__global__ __launch_bounds__(256) void tile_gemm_kernel(
        const _Float16* __restrict__ A,   // [M, 1024]
        const _Float16* __restrict__ Bt,  // [1024, 1024] row n = output col
        void* __restrict__ dstv,
        const float* __restrict__ bout) {
    __shared__ alignas(16) _Float16 shA[2][128 * LSTR];
    __shared__ alignas(16) _Float16 shB[2][128 * LSTR];

    int tid  = threadIdx.x;
    int wid  = tid >> 5, lane = tid & 31;
    int row0 = (blockIdx.x >> 3) * 128;   // 64 row blocks
    int col0 = (blockIdx.x & 7) * 128;    // 8 col blocks
    int ra = (wid >> 1) * 32;             // wave row offset in block tile
    int cb = (wid & 1) * 64;              // wave col offset in block tile

    v8f acc[2][4];
#pragma unroll
    for (int i = 0; i < 2; i++)
#pragma unroll
        for (int j = 0; j < 4; j++) acc[i][j] = zero8();

    // chunk c (0..511): 16B of a 128x32 panel; rr = c>>2, sub = c&3
    auto issue_copy = [&](int buf, int k0) {
#pragma unroll
        for (int i = 0; i < 2; i++) {
            int c  = tid + 256 * i;
            int rr = c >> 2, sub = c & 3;
            async_copy_b128(A  + (size_t)(row0 + rr) * DIM_ + k0 + sub * 8,
                            &shA[buf][rr * LSTR + sub * 8]);
            async_copy_b128(Bt + (size_t)(col0 + rr) * DIM_ + k0 + sub * 8,
                            &shB[buf][rr * LSTR + sub * 8]);
        }
    };

    issue_copy(0, 0);
    for (int step = 0; step < 32; ++step) {
        int cur = step & 1;
        if (step + 1 < 32) {
            issue_copy(cur ^ 1, (step + 1) * 32);
            asm volatile("s_wait_asynccnt 0x4" ::: "memory");  // current panel landed
        } else {
            asm volatile("s_wait_asynccnt 0x0" ::: "memory");
        }
        __syncthreads();  // all waves' panel writes visible

        const _Float16* sA = &shA[cur][0];
        const _Float16* sB = &shB[cur][0];
        v16h a0 = load_fragA(sA + (size_t)ra * LSTR, LSTR, lane);
        v16h a1 = load_fragA(sA + (size_t)(ra + 16) * LSTR, LSTR, lane);
#pragma unroll
        for (int j = 0; j < 4; j++) {
            v16h bf = load_fragB(sB + (size_t)(cb + j * 16) * LSTR, LSTR, lane);
            acc[0][j] = wmma_f16(a0, bf, acc[0][j]);
            acc[1][j] = wmma_f16(a1, bf, acc[1][j]);
        }
        __syncthreads();  // done reading before this buffer is refilled
    }

    int nn = lane & 15, hi4 = lane >> 4;
#pragma unroll
    for (int i = 0; i < 2; i++)
#pragma unroll
        for (int j = 0; j < 4; j++) {
            int gcc = col0 + cb + j * 16 + nn;
            float bb = (EPI == 2) ? bout[gcc] : 0.f;
#pragma unroll
            for (int r = 0; r < 8; r++) {
                int gr = row0 + ra + i * 16 + hi4 * 8 + r;  // token row qb*512+n
                if (EPI == 2) {
                    ((float*)dstv)[(size_t)gr * DIM_ + gcc] = acc[i][j][r] + bb;
                } else {
                    int h = gcc >> 7, d = gcc & 127, qb = gr >> 9, tn = gr & 511;
                    size_t addr;
                    if (EPI == 0)
                        addr = ((((size_t)h * QB_ + qb) * N_ + tn) * D_ + d);
                    else
                        addr = ((((size_t)h * QB_ + qb) * D_ + d) * N_ + tn);
                    ((_Float16*)dstv)[addr] = (_Float16)acc[i][j][r];
                }
            }
        }
}

// ---------------------------------------------------------------------------
// Row statistics for f_q / f_k (rows of 128): 1/norm, var-ratio, optional mean.
// ---------------------------------------------------------------------------
__global__ void row_stats_kernel(const _Float16* __restrict__ F,
                                 float* __restrict__ rnorm,
                                 float* __restrict__ ratio,
                                 float* __restrict__ mean_out) {
    int row  = blockIdx.x * 8 + (threadIdx.x >> 5);
    int lane = threadIdx.x & 31;
    const _Float16* p = F + (size_t)row * D_ + lane * 4;
    float s = 0.f, ss = 0.f;
#pragma unroll
    for (int i = 0; i < 4; i++) {
        float vv = (float)p[i];
        s += vv; ss += vv * vv;
    }
#pragma unroll
    for (int off = 16; off > 0; off >>= 1) {
        s  += __shfl_xor(s,  off, 32);
        ss += __shfl_xor(ss, off, 32);
    }
    if (lane == 0) {
        rnorm[row] = rsqrtf(ss);                                    // 1/||row||
        float var  = (ss - s * s * (1.f / D_)) * (1.f / (D_ - 1));  // ddof=1
        ratio[row] = 2.f * fminf(var, 1.f) / (var + 1.f);
        if (mean_out) mean_out[row] = s * (1.f / D_);
    }
}

// Per-head column means over 8192 rows: out[h][d]. One block per head.
__global__ void col_mean_kernel(const _Float16* __restrict__ F, float* __restrict__ out) {
    __shared__ float sh[256];
    int h = blockIdx.x, d = threadIdx.x & 127, half = threadIdx.x >> 7;
    const _Float16* p = F + ((size_t)h * ROWS_ + half * 4096) * D_ + d;
    float s = 0.f;
    for (int r = 0; r < 4096; r++) s += (float)p[(size_t)r * D_];
    sh[threadIdx.x] = s;
    __syncthreads();
    if (threadIdx.x < 128)
        out[h * D_ + d] = (sh[threadIdx.x] + sh[threadIdx.x + 128]) * (1.f / ROWS_);
}

// feat_corr off-diagonal sum of squares per head: S[h] += (C[d][e]/8192)^2, d!=e.
__global__ void corr_kernel(const _Float16* __restrict__ Fq,
                            const float* __restrict__ qmean,
                            float* __restrict__ S) {
    __shared__ float ta[16][17], tb[16][17];
    int h = blockIdx.x >> 6, tile = blockIdx.x & 63;
    int d0 = (tile >> 3) * 16, e0 = (tile & 7) * 16;
    int i = threadIdx.x >> 4, j = threadIdx.x & 15;
    const _Float16* base = Fq + (size_t)h * ROWS_ * D_;
    const float*    mb   = qmean + h * ROWS_;
    float acc = 0.f;
    for (int r0 = 0; r0 < ROWS_; r0 += 16) {
        int rr = r0 + i;
        float mu = mb[rr];
        ta[i][j] = (float)base[(size_t)rr * D_ + d0 + j] - mu;
        tb[i][j] = (float)base[(size_t)rr * D_ + e0 + j] - mu;
        __syncthreads();
#pragma unroll
        for (int r = 0; r < 16; r++) acc += ta[r][i] * tb[r][j];
        __syncthreads();
    }
    if (d0 + i != e0 + j) {
        float c = acc * (1.f / ROWS_);
        atomicAdd(&S[h], c * c);
    }
}

// Tiny weight-predictor MLP + softmax + decorr fold. Single block.
__global__ void predictor_kernel(const float* __restrict__ qg, const float* __restrict__ kg,
                                 const float* __restrict__ w1, const float* __restrict__ b1,
                                 const float* __restrict__ lng, const float* __restrict__ lnb,
                                 const float* __restrict__ w2, const float* __restrict__ b2,
                                 const float* __restrict__ S,
                                 float* __restrict__ alpha, float* __restrict__ wwv) {
    __shared__ float buf[1024];
    __shared__ float h1[1024];
    __shared__ float mu[8], rs[8], dsc[8], logits[24];
    int tid = threadIdx.x;
    for (int o = tid; o < 1024; o += 256) {
        int h = o >> 7, j = o & 127;
        const float* w = w1 + (size_t)j * 256;
        float s = b1[j];
        for (int i = 0; i < 128; i++) s += qg[h * 128 + i] * w[i];
        for (int i = 0; i < 128; i++) s += kg[h * 128 + i] * w[128 + i];
        buf[o] = s;
    }
    __syncthreads();
    if (tid < 8) {
        float s = 0.f, ss = 0.f;
        for (int j = 0; j < 128; j++) { float vv = buf[tid * 128 + j]; s += vv; ss += vv * vv; }
        float m = s * (1.f / 128.f);
        mu[tid] = m;
        rs[tid] = rsqrtf(ss * (1.f / 128.f) - m * m + 1e-5f);
        dsc[tid] = __expf(-5.f * sqrtf(S[tid]) * (1.f / (float)(D_ * D_)));
    }
    __syncthreads();
    for (int o = tid; o < 1024; o += 256) {
        int h = o >> 7, j = o & 127;
        h1[o] = fmaxf((buf[o] - mu[h]) * rs[h] * lng[j] + lnb[j], 0.f);
    }
    __syncthreads();
    if (tid < 24) {
        int h = tid / 3, c = tid % 3;
        float s = b2[c];
        for (int j = 0; j < 128; j++) s += h1[h * 128 + j] * w2[c * 128 + j];
        logits[tid] = s;
    }
    __syncthreads();
    if (tid < 8) {
        float l0 = logits[tid * 3], l1 = logits[tid * 3 + 1], l2 = logits[tid * 3 + 2];
        float mx = fmaxf(l0, fmaxf(l1, l2));
        float e0 = __expf(l0 - mx), e1 = __expf(l1 - mx), e2 = __expf(l2 - mx);
        float inv = 1.f / (e0 + e1 + e2);
        alpha[tid] = e0 * inv + e1 * inv * dsc[tid];
        wwv[tid]   = e2 * inv;
    }
}

// ---------------------------------------------------------------------------
// Score GEMM: per (h,qb) dots = f_q f_k^T [512x512, K=128], fused mixing:
//   mixed = dot*alpha[h]*(1/qn)(1/kn) + ww[h]*qr*kr  -> f16
// ---------------------------------------------------------------------------
__global__ __launch_bounds__(256) void score_gemm_kernel(
        const _Float16* __restrict__ fq16, const _Float16* __restrict__ fk16,
        const float* __restrict__ rqn, const float* __restrict__ rkn,
        const float* __restrict__ qr,  const float* __restrict__ kr,
        const float* __restrict__ alpha, const float* __restrict__ wwv,
        _Float16* __restrict__ mixed) {
    int task = blockIdx.x * 8 + (threadIdx.x >> 5);
    int lane = threadIdx.x & 31;
    int hq = task >> 7;
    int t  = task & 127;
    int row0 = (t >> 3) * 32;
    int col0 = (t & 7) * 64;
    int h = hq >> 4;

    const _Float16* A = fq16 + (size_t)hq * N_ * D_;
    const _Float16* B = fk16 + (size_t)hq * N_ * D_;

    v8f acc[2][4];
#pragma unroll
    for (int i = 0; i < 2; i++)
#pragma unroll
        for (int j = 0; j < 4; j++) acc[i][j] = zero8();

#pragma unroll
    for (int k0 = 0; k0 < D_; k0 += 32) {
        v16h a0 = load_fragA(A + (size_t)row0 * D_ + k0, D_, lane);
        v16h a1 = load_fragA(A + (size_t)(row0 + 16) * D_ + k0, D_, lane);
#pragma unroll
        for (int j = 0; j < 4; j++) {
            v16h bf = load_fragB(B + (size_t)(col0 + j * 16) * D_ + k0, D_, lane);
            acc[0][j] = wmma_f16(a0, bf, acc[0][j]);
            acc[1][j] = wmma_f16(a1, bf, acc[1][j]);
        }
    }

    const float* rq  = rqn + hq * N_;
    const float* qrr = qr  + hq * N_;
    const float* rk  = rkn + hq * N_;
    const float* krr = kr  + hq * N_;
    float al = alpha[h], ww = wwv[h];
    _Float16* mh = mixed + (size_t)hq * N_ * N_;

    int nn = lane & 15, hi4 = lane >> 4;
#pragma unroll
    for (int i = 0; i < 2; i++) {
        float s1[8], t1[8];
#pragma unroll
        for (int r = 0; r < 8; r++) {
            int gr = row0 + i * 16 + hi4 * 8 + r;
            s1[r] = al * rq[gr];
            t1[r] = ww * qrr[gr];
        }
#pragma unroll
        for (int j = 0; j < 4; j++) {
            int gc = col0 + j * 16 + nn;
            float s2 = rk[gc], t2 = krr[gc];
#pragma unroll
            for (int r = 0; r < 8; r++) {
                int gr = row0 + i * 16 + hi4 * 8 + r;
                mh[(size_t)gr * N_ + gc] = (_Float16)(acc[i][j][r] * s1[r] * s2 + t1[r] * t2);
            }
        }
    }
}

// ---------------------------------------------------------------------------
// Output einsum: per (h,qb)  O[512,128] = mixed[512,512] @ f_v[512,128]
// (B operand from pre-transposed fvt[h][qb][d][m]).  Stored to fo[r][h*128+d] f16.
// ---------------------------------------------------------------------------
__global__ __launch_bounds__(256) void out_gemm_kernel(
        const _Float16* __restrict__ mixed,
        const _Float16* __restrict__ fvt,
        _Float16* __restrict__ fo) {
    int task = blockIdx.x * 8 + (threadIdx.x >> 5);
    int lane = threadIdx.x & 31;
    int hq = task >> 5;
    int t  = task & 31;
    int row0 = (t >> 1) * 32;
    int col0 = (t & 1) * 64;
    int h = hq >> 4, qb = hq & 15;

    const _Float16* A  = mixed + (size_t)hq * N_ * N_;
    const _Float16* Bt = fvt   + (size_t)hq * D_ * N_;

    v8f acc[2][4];
#pragma unroll
    for (int i = 0; i < 2; i++)
#pragma unroll
        for (int j = 0; j < 4; j++) acc[i][j] = zero8();

    for (int k0 = 0; k0 < N_; k0 += 32) {
        v16h a0 = load_fragA(A + (size_t)row0 * N_ + k0, N_, lane);
        v16h a1 = load_fragA(A + (size_t)(row0 + 16) * N_ + k0, N_, lane);
#pragma unroll
        for (int j = 0; j < 4; j++) {
            v16h bf = load_fragB(Bt + (size_t)(col0 + j * 16) * N_ + k0, N_, lane);
            acc[0][j] = wmma_f16(a0, bf, acc[0][j]);
            acc[1][j] = wmma_f16(a1, bf, acc[1][j]);
        }
    }

    int nn = lane & 15, hi4 = lane >> 4;
#pragma unroll
    for (int i = 0; i < 2; i++)
#pragma unroll
        for (int j = 0; j < 4; j++)
#pragma unroll
            for (int r = 0; r < 8; r++) {
                int gr = row0 + i * 16 + hi4 * 8 + r;   // n within (h,qb)
                int gc = col0 + j * 16 + nn;            // d
                fo[((size_t)(qb * N_ + gr)) * DIM_ + h * D_ + gc] = (_Float16)acc[i][j][r];
            }
}

// ---------------------------------------------------------------------------
extern "C" void kernel_launch(void* const* d_in, const int* in_sizes, int n_in,
                              void* d_out, int out_size, void* d_ws, size_t ws_size,
                              hipStream_t stream) {
    (void)in_sizes; (void)n_in; (void)out_size; (void)ws_size;
    const float* q      = (const float*)d_in[0];
    const float* k      = (const float*)d_in[1];
    const float* v      = (const float*)d_in[2];
    const float* ln_g   = (const float*)d_in[3];
    const float* ln_b   = (const float*)d_in[4];
    const float* w_in   = (const float*)d_in[5];
    const float* wp_w1  = (const float*)d_in[6];
    const float* wp_b1  = (const float*)d_in[7];
    const float* wp_lng = (const float*)d_in[8];
    const float* wp_lnb = (const float*)d_in[9];
    const float* wp_w2  = (const float*)d_in[10];
    const float* wp_b2  = (const float*)d_in[11];
    const float* w_out  = (const float*)d_in[12];
    const float* b_out  = (const float*)d_in[13];
    float* out = (float*)d_out;

    char* ws = (char*)d_ws;
    size_t off = 0;
    auto alloc = [&](size_t bytes) -> void* {
        void* p = (void*)(ws + off);
        off = (off + bytes + 255) & ~(size_t)255;
        return p;
    };

    _Float16* xln16   = (_Float16*)alloc((size_t)ROWS_ * DIM_ * 2);
    _Float16* fq16    = (_Float16*)alloc((size_t)HQ_ * N_ * D_ * 2);
    _Float16* fk16    = (_Float16*)alloc((size_t)HQ_ * N_ * D_ * 2);
    _Float16* fvt16   = (_Float16*)alloc((size_t)HQ_ * D_ * N_ * 2);
    _Float16* mixed16 = (_Float16*)alloc((size_t)HQ_ * N_ * N_ * 2);
    _Float16* fo16    = (_Float16*)alloc((size_t)ROWS_ * DIM_ * 2);
    _Float16* win16   = (_Float16*)alloc((size_t)DIM_ * DIM_ * 2);
    _Float16* wout16  = (_Float16*)alloc((size_t)DIM_ * DIM_ * 2);
    float* rqn    = (float*)alloc((size_t)HQ_ * N_ * 4);
    float* rkn    = (float*)alloc((size_t)HQ_ * N_ * 4);
    float* qratio = (float*)alloc((size_t)HQ_ * N_ * 4);
    float* kratio = (float*)alloc((size_t)HQ_ * N_ * 4);
    float* qmean  = (float*)alloc((size_t)HQ_ * N_ * 4);
    float* qglob  = (float*)alloc((size_t)H_ * D_ * 4);
    float* kglob  = (float*)alloc((size_t)H_ * D_ * 4);
    float* Sacc   = (float*)alloc(256);
    float* alpha  = (float*)alloc(256);
    float* wwv    = (float*)alloc(256);

    // 0) zero the per-head corr accumulator (ws is poisoned; atomics need 0)
    zero_kernel<<<1, 32, 0, stream>>>(Sacc, 8);

    // 1) weights to f16 (row-major == B-operand layout for X @ W^T)
    cvt_f32_f16_kernel<<<4096, 256, 0, stream>>>(w_in,  win16,  DIM_ * DIM_);
    cvt_f32_f16_kernel<<<4096, 256, 0, stream>>>(w_out, wout16, DIM_ * DIM_);

    // 2) projections: LN -> f16, then async-LDS tiled WMMA GEMM, scatter store
    ln_rows_kernel<<<ROWS_, 256, 0, stream>>>(q, ln_g, ln_b, xln16);
    tile_gemm_kernel<0><<<512, 256, 0, stream>>>(xln16, win16, fq16, nullptr);
    ln_rows_kernel<<<ROWS_, 256, 0, stream>>>(k, ln_g, ln_b, xln16);
    tile_gemm_kernel<0><<<512, 256, 0, stream>>>(xln16, win16, fk16, nullptr);
    ln_rows_kernel<<<ROWS_, 256, 0, stream>>>(v, ln_g, ln_b, xln16);
    tile_gemm_kernel<1><<<512, 256, 0, stream>>>(xln16, win16, fvt16, nullptr);

    // 3) statistics
    row_stats_kernel<<<HQ_ * N_ / 8, 256, 0, stream>>>(fq16, rqn, qratio, qmean);
    row_stats_kernel<<<HQ_ * N_ / 8, 256, 0, stream>>>(fk16, rkn, kratio, nullptr);
    col_mean_kernel<<<H_, 256, 0, stream>>>(fq16, qglob);
    col_mean_kernel<<<H_, 256, 0, stream>>>(fk16, kglob);
    corr_kernel<<<H_ * 64, 256, 0, stream>>>(fq16, qmean, Sacc);
    predictor_kernel<<<1, 256, 0, stream>>>(qglob, kglob, wp_w1, wp_b1, wp_lng, wp_lnb,
                                            wp_w2, wp_b2, Sacc, alpha, wwv);

    // 4) dots + fused cosine/var/decorr mixing -> mixed f16
    score_gemm_kernel<<<2048, 256, 0, stream>>>(fq16, fk16, rqn, rkn, qratio, kratio,
                                                alpha, wwv, mixed16);

    // 5) mixed @ f_v  -> fo f16 (laid out [qb*N+n][h*D+d])
    out_gemm_kernel<<<512, 256, 0, stream>>>(mixed16, fvt16, fo16);

    // 6) final projection + bias -> f32 output
    tile_gemm_kernel<2><<<512, 256, 0, stream>>>(fo16, wout16, out, b_out);
}
